// Network_8349416423424
// MI455X (gfx1250) — matrix operand order repaired and checked by reading, not compiled, not run on hardware
//
#include <hip/hip_runtime.h>
#include <stdint.h>

// ---------------------------------------------------------------------------
// CDNA5 (gfx1250) dueling-DQN network. Heavy math: v_wmma_f32_16x16x32_bf16.
// Implicit-GEMM conv (geometry templated -> constant-div magic multiplies),
// TK=64 staged per barrier pair: 8 WMMAs per stage, all LDS traffic b128.
// ---------------------------------------------------------------------------

typedef __attribute__((ext_vector_type(16))) __bf16 v16bf;
typedef __attribute__((ext_vector_type(8)))  __bf16 v8bf;
typedef __attribute__((ext_vector_type(8)))  float  v8f;
typedef __attribute__((ext_vector_type(4)))  float  v4f;

__device__ __forceinline__ __bf16 f2bf(float f) {
    union { float f; uint32_t u; } in;
    in.f = f;
    uint32_t u = in.u;
    u += 0x7FFFu + ((u >> 16) & 1u);               // round-to-nearest-even
    union { unsigned short s; __bf16 b; } out;
    out.s = (unsigned short)(u >> 16);
    return out.b;
}

// ---------------------------------------------------------------------------
// Implicit-GEMM convolution, geometry as template parameters.
//   C[M=Cout, N=B*Ho*Wo] = relu( W[Cout,K] x P[K,N] + bias ),  K=Cin*KK*KK
// Block tile TMt x TNt (TMt*TNt == 8192), 8 waves, each wave 32x32 C.
// TK=64 per stage -> 8 v_wmma_f32_16x16x32_bf16 between barrier pairs.
// A tile [TMt][64] bf16, B tile transposed [TNt][64] bf16; rows padded to
// 72 elems (144B) so every fragment/staging access is a b128.
// ---------------------------------------------------------------------------
template <int Cin, int Hin, int Win, int Cout, int KK, int Ho, int Wo,
          int TMt, int TNt>
__global__ __launch_bounds__(256)
void conv_wmma(const float* __restrict__ in, const float* __restrict__ w,
               const float* __restrict__ bias, float* __restrict__ out,
               const float* __restrict__ pre_add)
{
    constexpr int Bn   = 128;
    constexpr int HW   = Ho * Wo;
    constexpr int N    = Bn * HW;
    constexpr int K    = Cin * KK * KK;
    constexpr int KHW  = KK * KK;
    constexpr int TK   = 64;
    constexpr int LDK  = 72;              // padded row stride (144B = 9*16B)
    constexpr int WN   = TNt / 32;        // waves along N
    constexpr int WM   = 8 / WN;          // waves along M
    static_assert(WM * 32 == TMt && TMt * TNt == 8192, "tile config");
    constexpr int A_PER_T = TMt * TK / 256;          // 16 (TMt=64) or 8 (TMt=32)
    constexpr int A_CPR   = TK / A_PER_T;            // chunks per A row
    constexpr int KPT     = TNt / 4;                 // K elems per thread for B

    const int n_base = blockIdx.x * TNt;
    const int m_base = blockIdx.y * TMt;
    const int tid  = threadIdx.x;
    const int lane = tid & 31;
    const int wid  = tid >> 5;
    const int wave_m = (wid / WN) * 32;
    const int wave_n = (wid % WN) * 32;
    const int half = lane >> 4;
    const int l16  = lane & 15;

    const float adder = pre_add ? *pre_add : 0.0f;

    __shared__ __bf16 As[TMt][LDK];       // [M][K]
    __shared__ __bf16 Bs[TNt][LDK];       // [N][K]  (transposed patch tile)

    // per-thread staging coordinates (loop invariant)
    const int arow = tid / A_CPR;
    const int akc  = (tid % A_CPR) * A_PER_T;
    const int agm  = m_base + arow;

    const int bn  = tid % TNt;
    const int bkg = (tid / TNt) * KPT;
    const int gn  = n_base + bn;
    // decompose n once (constant divisions -> magic multiplies)
    const int bb  = gn / HW;
    const int rr  = gn - bb * HW;
    const int oh  = rr / Wo;
    const int ow  = rr - oh * Wo;
    const bool nok = (gn < N);
    const float* inb = in + (bb * Cin * Hin + oh) * Win + ow;

    v8f acc[2][2];
    #pragma unroll
    for (int i = 0; i < 2; ++i)
        #pragma unroll
        for (int j = 0; j < 2; ++j)
            acc[i][j] = {};

    for (int k0 = 0; k0 < K; k0 += TK) {
        // ---- stage A (weights): A_PER_T contiguous K per thread ----
        {
            #pragma unroll
            for (int c = 0; c < A_PER_T / 8; ++c) {
                v8bf r;
                if (agm < Cout) {
                    v4f t0 = *(const v4f*)&w[agm * K + k0 + akc + c * 8];
                    v4f t1 = *(const v4f*)&w[agm * K + k0 + akc + c * 8 + 4];
                    #pragma unroll
                    for (int e = 0; e < 4; ++e) { r[e] = f2bf(t0[e]); r[4 + e] = f2bf(t1[e]); }
                } else {
                    #pragma unroll
                    for (int e = 0; e < 8; ++e) r[e] = f2bf(0.0f);
                }
                *(v8bf*)&As[arow][akc + c * 8] = r;
            }
            if (agm < Cout && k0 + TK < K)
                __builtin_prefetch(&w[agm * K + k0 + TK + akc], 0, 0);
        }
        // ---- stage B (patches): one n per thread, KPT contiguous k,
        //      (ci,kh,kw) advanced with carry chain (no division) ----
        {
            int gk = k0 + bkg;
            int ci = gk / KHW;
            int rm = gk - ci * KHW;
            int kh = rm / KK;
            int kw = rm - kh * KK;
            #pragma unroll
            for (int c = 0; c < KPT / 8; ++c) {
                v8bf r;
                #pragma unroll
                for (int e = 0; e < 8; ++e) {
                    float v = nok ? (inb[(ci * Hin + kh) * Win + kw] + adder) : 0.0f;
                    r[e] = f2bf(v);
                    if (++kw == KK) { kw = 0; if (++kh == KK) { kh = 0; ++ci; } }
                }
                *(v8bf*)&Bs[bn][bkg + c * 8] = r;
            }
        }
        __syncthreads();

        // ---- two K=32 sub-steps: 8 WMMAs between barrier pairs ----
        #pragma unroll
        for (int ks = 0; ks < 2; ++ks) {
            const int kb = ks * 32;
            v16bf a_frag[2], b_frag[2];
            #pragma unroll
            for (int t = 0; t < 2; ++t) {
                const int mrow = wave_m + t * 16 + l16;
                v8bf lo = *(const v8bf*)&As[mrow][kb + half * 8];
                v8bf hi = *(const v8bf*)&As[mrow][kb + 16 + half * 8];
                #pragma unroll
                for (int e = 0; e < 8; ++e) { a_frag[t][e] = lo[e]; a_frag[t][8 + e] = hi[e]; }
            }
            #pragma unroll
            for (int t = 0; t < 2; ++t) {
                const int ncol = wave_n + t * 16 + l16;
                v8bf b0 = *(const v8bf*)&Bs[ncol][kb + half * 8];
                v8bf b1 = *(const v8bf*)&Bs[ncol][kb + 16 + half * 8];
                #pragma unroll
                for (int e = 0; e < 8; ++e) { b_frag[t][e] = b0[e]; b_frag[t][8 + e] = b1[e]; }
            }
            #pragma unroll
            for (int i = 0; i < 2; ++i)
                #pragma unroll
                for (int j = 0; j < 2; ++j)
                    acc[i][j] = __builtin_amdgcn_wmma_f32_16x16x32_bf16(
                        false, a_frag[i], false, b_frag[j],
                        (short)0, acc[i][j], false, false);
        }
        __syncthreads();
    }

    // ---- epilogue: bias + relu, scatter to NCHW (coalesced across lanes) ----
    #pragma unroll
    for (int i = 0; i < 2; ++i) {
        #pragma unroll
        for (int j = 0; j < 2; ++j) {
            #pragma unroll
            for (int e = 0; e < 8; ++e) {
                const int gm = m_base + wave_m + i * 16 + e + half * 8;
                const int go = n_base + wave_n + j * 16 + l16;
                if (gm < Cout && go < N) {
                    float v = acc[i][j][e] + bias[gm];
                    v = v > 0.0f ? v : 0.0f;
                    const int b2  = go / HW;
                    const int r2  = go - b2 * HW;
                    const int oh2 = r2 / Wo;
                    const int ow2 = r2 - oh2 * Wo;
                    out[((b2 * Cout + gm) * Ho + oh2) * Wo + ow2] = v;
                }
            }
        }
    }
}

// ---------------------------------------------------------------------------
// WMMA fully-connected: out[n][m] = relu( X[n][:] . W[m][:] + bias[m] )
// Shapes are exact multiples of the tile (N=128, M%64==0, K%64==0): no guards.
// ---------------------------------------------------------------------------
__global__ __launch_bounds__(256)
void fc_wmma(const float* __restrict__ X,    // [N,K]
             const float* __restrict__ Wt,   // [M,K]
             const float* __restrict__ bias, // [M]
             float* __restrict__ out,        // [N,M]
             int M, int K)
{
    constexpr int TK  = 64;
    constexpr int LDK = 72;
    const int n_base = blockIdx.x * 128;
    const int m_base = blockIdx.y * 64;
    const int tid  = threadIdx.x;
    const int lane = tid & 31;
    const int wid  = tid >> 5;
    const int wave_m = (wid >> 2) * 32;
    const int wave_n = (wid & 3)  * 32;
    const int half = lane >> 4;
    const int l16  = lane & 15;

    __shared__ __bf16 As[64][LDK];
    __shared__ __bf16 Bs[128][LDK];

    const int arow = tid >> 2;            // 64 rows x 4 chunks of 16
    const int akc  = (tid & 3) * 16;
    const int agm  = m_base + arow;
    const int bn   = tid & 127;           // 128 rows x 2 chunks of 32
    const int bkg  = (tid >> 7) * 32;
    const int gn   = n_base + bn;

    v8f acc[2][2];
    #pragma unroll
    for (int i = 0; i < 2; ++i)
        #pragma unroll
        for (int j = 0; j < 2; ++j)
            acc[i][j] = {};

    for (int k0 = 0; k0 < K; k0 += TK) {
        {   // A: weights, 16 contiguous K
            #pragma unroll
            for (int c = 0; c < 2; ++c) {
                v4f t0 = *(const v4f*)&Wt[agm * K + k0 + akc + c * 8];
                v4f t1 = *(const v4f*)&Wt[agm * K + k0 + akc + c * 8 + 4];
                v8bf r;
                #pragma unroll
                for (int e = 0; e < 4; ++e) { r[e] = f2bf(t0[e]); r[4 + e] = f2bf(t1[e]); }
                *(v8bf*)&As[arow][akc + c * 8] = r;
            }
        }
        {   // B: activations (transposed tile), 32 contiguous K
            #pragma unroll
            for (int c = 0; c < 4; ++c) {
                v4f t0 = *(const v4f*)&X[gn * K + k0 + bkg + c * 8];
                v4f t1 = *(const v4f*)&X[gn * K + k0 + bkg + c * 8 + 4];
                v8bf r;
                #pragma unroll
                for (int e = 0; e < 4; ++e) { r[e] = f2bf(t0[e]); r[4 + e] = f2bf(t1[e]); }
                *(v8bf*)&Bs[bn][bkg + c * 8] = r;
            }
        }
        __syncthreads();

        #pragma unroll
        for (int ks = 0; ks < 2; ++ks) {
            const int kb = ks * 32;
            v16bf a_frag[2], b_frag[2];
            #pragma unroll
            for (int t = 0; t < 2; ++t) {
                const int mrow = wave_m + t * 16 + l16;
                v8bf lo = *(const v8bf*)&As[mrow][kb + half * 8];
                v8bf hi = *(const v8bf*)&As[mrow][kb + 16 + half * 8];
                #pragma unroll
                for (int e = 0; e < 8; ++e) { a_frag[t][e] = lo[e]; a_frag[t][8 + e] = hi[e]; }
            }
            #pragma unroll
            for (int t = 0; t < 2; ++t) {
                const int ncol = wave_n + t * 16 + l16;
                v8bf b0 = *(const v8bf*)&Bs[ncol][kb + half * 8];
                v8bf b1 = *(const v8bf*)&Bs[ncol][kb + 16 + half * 8];
                #pragma unroll
                for (int e = 0; e < 8; ++e) { b_frag[t][e] = b0[e]; b_frag[t][8 + e] = b1[e]; }
            }
            #pragma unroll
            for (int i = 0; i < 2; ++i)
                #pragma unroll
                for (int j = 0; j < 2; ++j)
                    acc[i][j] = __builtin_amdgcn_wmma_f32_16x16x32_bf16(
                        false, a_frag[i], false, b_frag[j],
                        (short)0, acc[i][j], false, false);
        }
        __syncthreads();
    }

    #pragma unroll
    for (int i = 0; i < 2; ++i)
        #pragma unroll
        for (int j = 0; j < 2; ++j)
            #pragma unroll
            for (int e = 0; e < 8; ++e) {
                const int gm = m_base + wave_m + i * 16 + e + half * 8;
                const int go = n_base + wave_n + j * 16 + l16;
                float v = acc[i][j][e] + bias[gm];
                out[go * M + gm] = v > 0.0f ? v : 0.0f;
            }
}

// ---------------------------------------------------------------------------
// The reference only consumes relu(fc1)[0, 30] as a broadcast scalar.
// ---------------------------------------------------------------------------
__global__ void fc1_scalar_kernel(const float* __restrict__ g,
                                  const float* __restrict__ v,
                                  const float* __restrict__ w,  // [67,4]
                                  const float* __restrict__ b,
                                  float* __restrict__ out_scalar, int idx)
{
    if (threadIdx.x == 0 && blockIdx.x == 0) {
        float s = b[idx];
        s += g[0] * w[idx * 4 + 0] + g[1] * w[idx * 4 + 1];
        s += v[0] * w[idx * 4 + 2] + v[1] * w[idx * 4 + 3];
        out_scalar[0] = s > 0.0f ? s : 0.0f;
    }
}

// ---------------------------------------------------------------------------
// Dueling head: out = adv + val - mean(adv).  One wave32 per batch row.
// ---------------------------------------------------------------------------
__global__ __launch_bounds__(32)
void head_kernel(const float* __restrict__ x,    // [Bn,Kd]
                 const float* __restrict__ ea_w, // [Od,Kd]
                 const float* __restrict__ ea_b,
                 const float* __restrict__ ev_w, // [1,Kd]
                 const float* __restrict__ ev_b,
                 float* __restrict__ out,        // [Bn,Od]
                 int Kd, int Od)
{
    const int b    = blockIdx.x;
    const int lane = threadIdx.x;
    const float* xb = x + b * Kd;

    float adv[16];
    float asum = 0.0f;
    for (int o = 0; o < Od; ++o) {
        float p = 0.0f;
        for (int k = lane; k < Kd; k += 32) p += xb[k] * ea_w[o * Kd + k];
        #pragma unroll
        for (int off = 16; off > 0; off >>= 1) p += __shfl_xor(p, off, 32);
        adv[o] = p + ea_b[o];
        asum  += adv[o];
    }
    float val = 0.0f;
    for (int k = lane; k < Kd; k += 32) val += xb[k] * ev_w[k];
    #pragma unroll
    for (int off = 16; off > 0; off >>= 1) val += __shfl_xor(val, off, 32);
    val += ev_b[0];

    if (lane == 0) {
        float mean = asum / (float)Od;
        for (int o = 0; o < Od; ++o)
            out[b * Od + o] = adv[o] + val - mean;
    }
}

// ---------------------------------------------------------------------------
extern "C" void kernel_launch(void* const* d_in, const int* in_sizes, int n_in,
                              void* d_out, int out_size, void* d_ws, size_t ws_size,
                              hipStream_t stream)
{
    (void)in_sizes; (void)n_in; (void)out_size; (void)ws_size;

    const float* state_m = (const float*)d_in[0];
    const float* state_g = (const float*)d_in[1];
    const float* state_v = (const float*)d_in[2];
    const float* c1w = (const float*)d_in[3];  const float* c1b = (const float*)d_in[4];
    const float* c2w = (const float*)d_in[5];  const float* c2b = (const float*)d_in[6];
    const float* c3w = (const float*)d_in[7];  const float* c3b = (const float*)d_in[8];
    const float* c4w = (const float*)d_in[9];  const float* c4b = (const float*)d_in[10];
    const float* f1w = (const float*)d_in[11]; const float* f1b = (const float*)d_in[12];
    const float* f2w = (const float*)d_in[13]; const float* f2b = (const float*)d_in[14];
    const float* f3w = (const float*)d_in[15]; const float* f3b = (const float*)d_in[16];
    const float* eaw = (const float*)d_in[17]; const float* eab = (const float*)d_in[18];
    const float* evw = (const float*)d_in[19]; const float* evb = (const float*)d_in[20];

    // workspace layout (floats): two ping-pong activation slabs + FC tails
    float* ws    = (float*)d_ws;
    float* slabA = ws;                                  // 19,501,056 (conv1 out)
    float* slabB = slabA + 19501056;                    // 11,829,248 (conv2 out)
    float* fc2o  = slabB + 11829248;                    // 65,536
    float* fc3o  = fc2o + 65536;                        // 65,536
    float* scal  = fc3o + 65536;                        // 1

    dim3 blk(256);

    // conv1: [128,4,100,100] -> [128,32,69,69]   (32x256 tile, N=609408)
    conv_wmma<4, 100, 100, 32, 32, 69, 69, 32, 256>
        <<<dim3((609408 + 255) / 256, 1), blk, 0, stream>>>(state_m, c1w, c1b, slabA, nullptr);
    // conv2: -> [128,64,38,38]   N=184832
    conv_wmma<32, 69, 69, 64, 32, 38, 38, 64, 128>
        <<<dim3(184832 / 128, 1), blk, 0, stream>>>(slabA, c2w, c2b, slabB, nullptr);
    // conv3: -> [128,64,31,31]   N=123008
    conv_wmma<64, 38, 38, 64, 8, 31, 31, 64, 128>
        <<<dim3(123008 / 128, 1), blk, 0, stream>>>(slabB, c3w, c3b, slabA, nullptr);

    fc1_scalar_kernel<<<dim3(1), dim3(32), 0, stream>>>(state_g, state_v,
                                                        f1w, f1b, scal, 30);

    // conv4 chain (first application adds the broadcast scalar)
    conv_wmma<64, 31, 31, 64, 8, 24, 24, 64, 128>
        <<<dim3(73728 / 128, 1), blk, 0, stream>>>(slabA, c4w, c4b, slabB, scal);
    conv_wmma<64, 24, 24, 64, 8, 17, 17, 64, 128>
        <<<dim3(36992 / 128, 1), blk, 0, stream>>>(slabB, c4w, c4b, slabA, nullptr);
    conv_wmma<64, 17, 17, 64, 8, 10, 10, 64, 128>
        <<<dim3(12800 / 128, 1), blk, 0, stream>>>(slabA, c4w, c4b, slabB, nullptr);

    // fc2: [128,6400] x [512,6400]^T ; fc3: [128,512] x [512,512]^T
    fc_wmma<<<dim3(1, 512 / 64), blk, 0, stream>>>(slabB, f2w, f2b, fc2o, 512, 6400);
    fc_wmma<<<dim3(1, 512 / 64), blk, 0, stream>>>(fc2o, f3w, f3b, fc3o, 512, 512);

    head_kernel<<<dim3(128), dim3(32), 0, stream>>>(fc3o, eaw, eab, evw, evb,
                                                    (float*)d_out, 512, 9);
}
